// Head_13091060318559
// MI455X (gfx1250) — matrix operand order repaired and checked by reading, not compiled, not run on hardware
//
#include <hip/hip_runtime.h>
#include <hip/hip_bf16.h>
#include <cstdint>

typedef __attribute__((ext_vector_type(16))) _Float16 v16h;
typedef __attribute__((ext_vector_type(8)))  _Float16 v8h;
typedef __attribute__((ext_vector_type(8)))  float    v8f;

#define BATCH 8
#define SEQ   2048
#define EMB   1024
#define NH    64
#define ROWS  (BATCH*SEQ)   // 16384

__device__ __forceinline__ v8f wmma16(v16h a, v16h b, v8f c) {
  // D = A(16x32 f16) * B(32x16 f16) + C(16x16 f32)
  return __builtin_amdgcn_wmma_f32_16x16x32_f16(false, a, false, b, (short)0, c,
                                                false, false);
}

__device__ __forceinline__ v16h cat8(v8h lo, v8h hi) {
  v16h r;
#pragma unroll
  for (int i = 0; i < 8; ++i) { r[i] = lo[i]; r[8 + i] = hi[i]; }
  return r;
}

// ---------------------------------------------------------------------------
// Kernel 0: weight transpose + f32->f16 convert.
// Wt[w][h][c] = (f16) W_w[c][h]   (h-major so B-operand loads are contiguous)
// ---------------------------------------------------------------------------
__global__ void cvt_weights(const float* __restrict__ Wk,
                            const float* __restrict__ Wq,
                            const float* __restrict__ Wv,
                            _Float16* __restrict__ Wt) {
  int idx = blockIdx.x * blockDim.x + threadIdx.x;   // 3*64*1024 threads
  int w = idx >> 16;
  int r = idx & 0xFFFF;
  int h = r >> 10;
  int c = r & 1023;
  const float* W = (w == 0) ? Wk : (w == 1) ? Wq : Wv;
  Wt[idx] = (_Float16)W[c * NH + h];
}

// ---------------------------------------------------------------------------
// Kernel 1: projections K = x*Wk, Q = x*Wq, V = x*Wv via f16 WMMA.
// One wave handles a 16-row tile for one of the three matrices.
// Software-pipelined: operands for k0+32 are fetched before k0's WMMAs issue.
// K,Q stored row-major f16 [ROWS][64]; V stored transposed f16 [B][64][T].
// ---------------------------------------------------------------------------
__global__ void __launch_bounds__(256) proj_kernel(
    const float* __restrict__ x, const _Float16* __restrict__ Wt,
    _Float16* __restrict__ Kh, _Float16* __restrict__ Qh,
    _Float16* __restrict__ Vt) {
  const int lane = threadIdx.x & 31;
  const int wid  = blockIdx.x * 8 + (threadIdx.x >> 5);   // 0..3071
  const int w    = wid >> 10;            // 0=K, 1=Q, 2=V
  const int tile = wid & 1023;           // 16-row tile index
  const int m0   = tile << 4;
  const int half = lane >> 4;            // lane group
  const int l16  = lane & 15;
  const int aoff = half * 16;            // A-operand K-chunk offset
  const int boff = half * 16;            // B-operand K offset

  const float*    xrow = x + (size_t)(m0 + l16) * EMB;
  const _Float16* wb   = Wt + (size_t)w * NH * EMB + boff;

  auto loadA = [&](int k0) -> v16h {
    float4 f0 = *(const float4*)(xrow + k0 + aoff);
    float4 f1 = *(const float4*)(xrow + k0 + aoff + 4);
    float4 f2 = *(const float4*)(xrow + k0 + aoff + 8);
    float4 f3 = *(const float4*)(xrow + k0 + aoff + 12);
    v16h a;
    a[0]=(_Float16)f0.x;  a[1]=(_Float16)f0.y;  a[2]=(_Float16)f0.z;  a[3]=(_Float16)f0.w;
    a[4]=(_Float16)f1.x;  a[5]=(_Float16)f1.y;  a[6]=(_Float16)f1.z;  a[7]=(_Float16)f1.w;
    a[8]=(_Float16)f2.x;  a[9]=(_Float16)f2.y;  a[10]=(_Float16)f2.z; a[11]=(_Float16)f2.w;
    a[12]=(_Float16)f3.x; a[13]=(_Float16)f3.y; a[14]=(_Float16)f3.z; a[15]=(_Float16)f3.w;
    return a;
  };
  auto loadB = [&](int nt, int k0) -> v16h {
    return *(const v16h*)(wb + (size_t)(nt * 16 + l16) * EMB + k0);
  };

  v8f acc[4] = {};

  // prologue
  v16h a  = loadA(0);
  v16h b0 = loadB(0, 0), b1 = loadB(1, 0), b2 = loadB(2, 0), b3 = loadB(3, 0);

  for (int k0 = 32; k0 < EMB; k0 += 32) {
    // fetch next iteration's operands first (overlaps with WMMAs below)
    v16h an = loadA(k0);
    v16h c0 = loadB(0, k0), c1 = loadB(1, k0), c2 = loadB(2, k0), c3 = loadB(3, k0);
    acc[0] = wmma16(a, b0, acc[0]);
    acc[1] = wmma16(a, b1, acc[1]);
    acc[2] = wmma16(a, b2, acc[2]);
    acc[3] = wmma16(a, b3, acc[3]);
    a = an; b0 = c0; b1 = c1; b2 = c2; b3 = c3;
  }
  // epilogue
  acc[0] = wmma16(a, b0, acc[0]);
  acc[1] = wmma16(a, b1, acc[1]);
  acc[2] = wmma16(a, b2, acc[2]);
  acc[3] = wmma16(a, b3, acc[3]);

  if (w < 2) {
    _Float16* base = (w == 0) ? Kh : Qh;
#pragma unroll
    for (int nt = 0; nt < 4; ++nt) {
      int col = nt * 16 + l16;
#pragma unroll
      for (int e = 0; e < 8; ++e) {
        int rr = m0 + e + half * 8;
        base[(size_t)rr * NH + col] = (_Float16)acc[nt][e];
      }
    }
  } else {
    // V transposed: Vt[b][h][t]; consecutive e -> consecutive t => 16B store.
    int b  = m0 >> 11;
    int t0 = (m0 & (SEQ - 1)) + half * 8;
#pragma unroll
    for (int nt = 0; nt < 4; ++nt) {
      int col = nt * 16 + l16;
      v8h pk;
#pragma unroll
      for (int e = 0; e < 8; ++e) pk[e] = (_Float16)acc[nt][e];
      *(v8h*)(Vt + ((size_t)(b * NH + col) * SEQ + t0)) = pk;
    }
  }
}

// ---------------------------------------------------------------------------
// Kernel 2: flash attention. One wave per 16-row i-tile; j-blocks of 32.
// scores[i][j] = k[i].q[j] * 1/32 (reference quirk: k rows, q cols).
// Tiles remapped wave*128+block so each workgroup gets a uniform causal mix.
// ---------------------------------------------------------------------------
__global__ void __launch_bounds__(256) attn_kernel(
    const _Float16* __restrict__ Kh, const _Float16* __restrict__ Qh,
    const _Float16* __restrict__ Vt, float* __restrict__ out) {
  __shared__ _Float16 Psh[8][16 * 32];   // per-wave P staging (D-layout -> A-layout)
  const int lane = threadIdx.x & 31;
  const int wv   = threadIdx.x >> 5;
  const int tile = wv * 128 + blockIdx.x;  // 0..1023, balanced across blocks
  const int bi   = tile >> 7;              // batch (128 tiles per batch)
  const int i0g  = tile << 4;              // flattened global row
  const int i0   = i0g & (SEQ - 1);        // row within batch
  const int half = lane >> 4;
  const int l16  = lane & 15;
  const int aoff = half * 16;
  const int boff = half * 16;

  // K i-tile as two A operands (head dims 0..31 and 32..63), loaded once.
  const _Float16* krow = Kh + (size_t)(i0g + l16) * NH;
  v16h A0 = cat8(*(const v8h*)(krow + aoff),      *(const v8h*)(krow + 8 + aoff));
  v16h A1 = cat8(*(const v8h*)(krow + 32 + aoff), *(const v8h*)(krow + 40 + aoff));

  const _Float16* qbase = Qh + (size_t)bi * SEQ * NH;
  const _Float16* vbase = Vt + (size_t)bi * NH * SEQ;
  _Float16* P = &Psh[wv][0];

  v8f O[4] = {};
  float m_i[8], l_i[8];
#pragma unroll
  for (int e = 0; e < 8; ++e) { m_i[e] = -3.0e38f; l_i[e] = 0.0f; }

  const float sc = 0.03125f;   // 1024^-0.5

  for (int jb = 0; jb <= i0 + 15; jb += 32) {
    const _Float16* q0 = qbase + (size_t)(jb + l16) * NH + boff;
    const _Float16* q1 = qbase + (size_t)(jb + 16 + l16) * NH + boff;
    const _Float16* vb = vbase + jb + boff;

    // ---- scores: two 16x16 tiles (j = jb..jb+15, jb+16..jb+31) ----
    v8f S0 = {}, S1 = {};
    S0 = wmma16(A0, *(const v16h*)(q0),      S0);
    S0 = wmma16(A1, *(const v16h*)(q0 + 32), S0);
    S1 = wmma16(A0, *(const v16h*)(q1),      S1);
    S1 = wmma16(A1, *(const v16h*)(q1 + 32), S1);

    // V B-operands: independent of scores; issue now so the softmax VALU
    // chain below hides their latency.
    v16h Bv0 = *(const v16h*)(vb + (size_t)(0 * 16 + l16) * SEQ);
    v16h Bv1 = *(const v16h*)(vb + (size_t)(1 * 16 + l16) * SEQ);
    v16h Bv2 = *(const v16h*)(vb + (size_t)(2 * 16 + l16) * SEQ);
    v16h Bv3 = *(const v16h*)(vb + (size_t)(3 * 16 + l16) * SEQ);

    // prefetch next j-step's lines (speculative; dropped if out of range)
    __builtin_prefetch(q0 + 32 * NH, 0, 3);
    __builtin_prefetch(q1 + 32 * NH, 0, 3);
    __builtin_prefetch(vb + 32, 0, 3);

    const bool need_mask = (jb + 31 > i0);
#pragma unroll
    for (int e = 0; e < 8; ++e) {
      const int irow = i0 + e + half * 8;
      float s0 = S0[e] * sc;
      float s1 = S1[e] * sc;
      if (need_mask) {
        s0 = (jb + l16      > irow) ? -3.0e38f : s0;
        s1 = (jb + 16 + l16 > irow) ? -3.0e38f : s1;
      }
      // row max across the 16 lanes holding this row
      float mx = fmaxf(s0, s1);
#pragma unroll
      for (int d = 1; d < 16; d <<= 1) mx = fmaxf(mx, __shfl_xor(mx, d, 32));
      const float mnew = fmaxf(m_i[e], mx);
      const float p0   = __expf(s0 - mnew);
      const float p1   = __expf(s1 - mnew);
      const float corr = __expf(m_i[e] - mnew);
      m_i[e] = mnew;
      float ps = p0 + p1;
#pragma unroll
      for (int d = 1; d < 16; d <<= 1) ps += __shfl_xor(ps, d, 32);
      l_i[e] = l_i[e] * corr + ps;
      O[0][e] *= corr; O[1][e] *= corr; O[2][e] *= corr; O[3][e] *= corr;
      // stage P (f16) row-major [16][32] in LDS
      const int rloc = e + half * 8;
      P[rloc * 32 + l16]      = (_Float16)p0;
      P[rloc * 32 + 16 + l16] = (_Float16)p1;
    }

    // per-wave LDS ordering: DS ops in-order within wave; wait for stores
    asm volatile("s_wait_dscnt 0" ::: "memory");

    // P as A operand (16x32); V^T columns already in registers
    const _Float16* prow = P + l16 * 32 + aoff;
    v16h Ap = cat8(*(const v8h*)(prow), *(const v8h*)(prow + 8));
    O[0] = wmma16(Ap, Bv0, O[0]);
    O[1] = wmma16(Ap, Bv1, O[1]);
    O[2] = wmma16(Ap, Bv2, O[2]);
    O[3] = wmma16(Ap, Bv3, O[3]);
  }

  // normalize and store fp32 output [ROWS][64]
  float* obase = out + (size_t)i0g * NH;
#pragma unroll
  for (int e = 0; e < 8; ++e) {
    const float rinv = 1.0f / l_i[e];
    const int rr = e + half * 8;
#pragma unroll
    for (int nt = 0; nt < 4; ++nt) {
      obase[(size_t)rr * NH + nt * 16 + l16] = O[nt][e] * rinv;
    }
  }
}

// ---------------------------------------------------------------------------
extern "C" void kernel_launch(void* const* d_in, const int* in_sizes, int n_in,
                              void* d_out, int out_size, void* d_ws, size_t ws_size,
                              hipStream_t stream) {
  const float* x  = (const float*)d_in[0];
  const float* Wk = (const float*)d_in[1];
  const float* Wq = (const float*)d_in[2];
  const float* Wv = (const float*)d_in[3];
  float* out = (float*)d_out;

  _Float16* ws = (_Float16*)d_ws;
  _Float16* Wt = ws;                               // 3*64*1024 halves
  _Float16* Kh = Wt + (size_t)3 * NH * EMB;        // ROWS*64
  _Float16* Qh = Kh + (size_t)ROWS * NH;           // ROWS*64
  _Float16* Vt = Qh + (size_t)ROWS * NH;           // B*64*T

  cvt_weights<<<dim3((3 * NH * EMB) / 256), dim3(256), 0, stream>>>(Wk, Wq, Wv, Wt);
  proj_kernel<<<dim3(384), dim3(256), 0, stream>>>(x, Wt, Kh, Qh, Vt);
  attn_kernel<<<dim3(128), dim3(256), 0, stream>>>(Kh, Qh, Vt, out);
}
